// PAAE_73237782332006
// MI455X (gfx1250) — matrix operand order repaired and checked by reading, not compiled, not run on hardware
//
#include <hip/hip_runtime.h>

typedef __attribute__((ext_vector_type(16))) _Float16 v16h;
typedef __attribute__((ext_vector_type(8)))  float    v8f;

#define B_ 2048
#define G_ 20000
#define P_ 512
#define L_ 200
#define H_ 128
#define E_ 64

static __device__ __forceinline__ v8f wmma16(v16h a, v16h b, v8f c) {
  // D = A(16x32 f16) * B(32x16 f16) + C(16x16 f32)
  return __builtin_amdgcn_wmma_f32_16x16x32_f16(false, a, false, b, (short)0, c,
                                                false, false);
}

static __device__ __forceinline__ v8f zero8() {
  v8f z = {0.f, 0.f, 0.f, 0.f, 0.f, 0.f, 0.f, 0.f};
  return z;
}

// A fragment 16x32 (MxK) from row-major f16 [rows][ldk].
// Contiguous k order (matches B): lane<16 -> M=lane, K=0..15; lane>=16 -> K=16..31.
static __device__ __forceinline__ v16h load_a_frag(const _Float16* A, int ldk,
                                                   int m0, int k0, int lane) {
  int m   = m0 + (lane & 15);
  int hi8 = (lane >> 4) * 16;
  const _Float16* p = A + (size_t)m * ldk + k0 + hi8;
  v16h a;
#pragma unroll
  for (int v = 0; v < 16; ++v) a[v] = p[v];
  return a;
}

// B fragment 32x16 (KxN) from N-major transposed storage Wt[n][k], ld=ldk.
// Contiguous k order (matches A): lane<16 -> N=lane, K=0..15 (2 per VGPR); lane>=16 -> K=16..31.
static __device__ __forceinline__ v16h load_b_frag(const _Float16* Wt, int ldk,
                                                   int n0, int k0, int lane) {
  int n    = n0 + (lane & 15);
  int hi16 = (lane >> 4) * 16;
  const _Float16* p = Wt + (size_t)n * ldk + k0 + hi16;
  v16h b;
#pragma unroll
  for (int v = 0; v < 16; ++v) b[v] = p[v];
  return b;
}

// ---------------------------------------------------------------------------
// Kernel 1: convert weights to f16, transposed to N-major [N][K] layouts.
// ---------------------------------------------------------------------------
__global__ __launch_bounds__(256) void convert_weights(
    const float* __restrict__ w1, const float* __restrict__ w2,
    const float* __restrict__ d1, const float* __restrict__ dw2,
    _Float16* __restrict__ w1t, _Float16* __restrict__ w2t,
    _Float16* __restrict__ d1t, _Float16* __restrict__ w2dt) {
  const int N1 = H_ * P_;        // w1t[128][512]  <- enc_w1(512x128)
  const int N2 = E_ * H_;        // w2t[64][128]   <- enc_w2(128x64)
  const int N3 = H_ * E_;        // d1t[128][64]   <- dec_w1(64x128)
  const int N4 = G_ * H_;        // w2dt[20000][128] <- dec_w2(128x20000)
  int i = blockIdx.x * 256 + threadIdx.x;
  if (i < N1) { int n = i / P_, k = i % P_; w1t[i] = (_Float16)w1[k * H_ + n]; return; }
  i -= N1;
  if (i < N2) { int n = i / H_, k = i % H_; w2t[i] = (_Float16)w2[k * E_ + n]; return; }
  i -= N2;
  if (i < N3) { int n = i / E_, k = i % E_; d1t[i] = (_Float16)d1[k * H_ + n]; return; }
  i -= N3;
  if (i < N4) { int n = i / H_, k = i % H_; w2dt[i] = (_Float16)dw2[(size_t)k * G_ + n]; }
}

// ---------------------------------------------------------------------------
// Kernel 2: transpose x (B x G) -> x_T (G x B) so the pathway gather is
// coalesced (lanes vary over b => contiguous addresses in x_T).
// ---------------------------------------------------------------------------
__global__ __launch_bounds__(256) void transpose_x(const float* __restrict__ x,
                                                   float* __restrict__ xt) {
  __shared__ float tile[32][33];
  int g0 = blockIdx.x * 32;
  int b0 = blockIdx.y * 32;
  int tx = threadIdx.x;  // 0..31
  int ty = threadIdx.y;  // 0..7
#pragma unroll
  for (int i = 0; i < 4; ++i)
    tile[ty + i * 8][tx] = x[(size_t)(b0 + ty + i * 8) * G_ + g0 + tx];
  __syncthreads();
#pragma unroll
  for (int i = 0; i < 4; ++i)
    xt[(size_t)(g0 + ty + i * 8) * B_ + b0 + tx] = tile[tx][ty + i * 8];
}

// ---------------------------------------------------------------------------
// Kernel 3: pathway gather + weighted sum + tanh -> a_h (f16, [B][P]).
// grid = (P, B/256), block = 256. One pathway per block.x.
// ---------------------------------------------------------------------------
__global__ __launch_bounds__(256) void pathway_kernel(
    const float* __restrict__ x, const float* __restrict__ xt,
    const int* __restrict__ pidx, const float* __restrict__ pw_w,
    const float* __restrict__ pw_b, _Float16* __restrict__ a_h, int use_xt) {
  __shared__ int   sidx[L_];
  __shared__ float sw[L_];
  int p = blockIdx.x;
  int t = threadIdx.x;
  if (t < L_) {
    sidx[t] = pidx[p * L_ + t];
    sw[t]   = pw_w[p * L_ + t];
  }
  __syncthreads();
  int b = blockIdx.y * 256 + t;
  float acc = 0.f;
  if (use_xt) {
#pragma unroll 4
    for (int l = 0; l < L_; ++l)
      acc += xt[(size_t)sidx[l] * B_ + b] * sw[l];   // coalesced across lanes
  } else {
    const float* xr = x + (size_t)b * G_;
#pragma unroll 4
    for (int l = 0; l < L_; ++l)
      acc += xr[sidx[l]] * sw[l];
  }
  a_h[(size_t)b * P_ + p] = (_Float16)tanhf(acc + pw_b[p]);
}

// ---------------------------------------------------------------------------
// Kernel 4: fused MLP  a(2048x512) ->relu h1(x128) -> z(x64) ->relu h2(x128)
// block = 128 threads (4 waves), each wave owns 16 rows; grid = 2048/64 = 32.
// Intermediates pass through per-wave LDS tiles in f16.
// ---------------------------------------------------------------------------
__global__ __launch_bounds__(128) void mlp_kernel(
    const _Float16* __restrict__ a_h, const _Float16* __restrict__ w1t,
    const float* __restrict__ b1, const _Float16* __restrict__ w2t,
    const float* __restrict__ b2, const _Float16* __restrict__ d1t,
    const float* __restrict__ db1, float* __restrict__ z_out,
    _Float16* __restrict__ h2h) {
  __shared__ _Float16 h1buf[4][16 * H_];  // 16 KB
  __shared__ _Float16 zbuf[4][16 * E_];   //  8 KB
  int lane = threadIdx.x & 31;
  int w    = threadIdx.x >> 5;
  int m0   = blockIdx.x * 64 + w * 16;
  int ncol = lane & 15;
  int mb   = (lane >> 4) * 8;

  // ---- stage 1: h1 = relu(a @ W1 + b1)   M=16, N=128, K=512
  v8f acc[8];
#pragma unroll
  for (int nt = 0; nt < 8; ++nt) acc[nt] = zero8();
  for (int kt = 0; kt < 16; ++kt) {
    v16h a = load_a_frag(a_h, P_, m0, kt * 32, lane);
#pragma unroll
    for (int nt = 0; nt < 8; ++nt) {
      v16h b = load_b_frag(w1t, P_, nt * 16, kt * 32, lane);
      acc[nt] = wmma16(a, b, acc[nt]);
    }
  }
#pragma unroll
  for (int nt = 0; nt < 8; ++nt) {
    float bias = b1[nt * 16 + ncol];
#pragma unroll
    for (int r = 0; r < 8; ++r) {
      float v = acc[nt][r] + bias;
      v = v > 0.f ? v : 0.f;
      h1buf[w][(mb + r) * H_ + nt * 16 + ncol] = (_Float16)v;
    }
  }
  __syncthreads();

  // ---- stage 2: z = h1 @ W2 + b2   M=16, N=64, K=128
  v8f zacc[4];
#pragma unroll
  for (int nt = 0; nt < 4; ++nt) zacc[nt] = zero8();
  for (int kt = 0; kt < 4; ++kt) {
    v16h a = load_a_frag(&h1buf[w][0], H_, 0, kt * 32, lane);
#pragma unroll
    for (int nt = 0; nt < 4; ++nt) {
      v16h b = load_b_frag(w2t, H_, nt * 16, kt * 32, lane);
      zacc[nt] = wmma16(a, b, zacc[nt]);
    }
  }
#pragma unroll
  for (int nt = 0; nt < 4; ++nt) {
    float bias = b2[nt * 16 + ncol];
#pragma unroll
    for (int r = 0; r < 8; ++r) {
      float v = zacc[nt][r] + bias;
      int m = m0 + mb + r;
      z_out[(size_t)m * E_ + nt * 16 + ncol] = v;       // output 1 (f32)
      zbuf[w][(mb + r) * E_ + nt * 16 + ncol] = (_Float16)v;
    }
  }
  __syncthreads();

  // ---- stage 3: h2 = relu(z @ D1 + db1)   M=16, N=128, K=64
  v8f hacc[8];
#pragma unroll
  for (int nt = 0; nt < 8; ++nt) hacc[nt] = zero8();
  for (int kt = 0; kt < 2; ++kt) {
    v16h a = load_a_frag(&zbuf[w][0], E_, 0, kt * 32, lane);
#pragma unroll
    for (int nt = 0; nt < 8; ++nt) {
      v16h b = load_b_frag(d1t, E_, nt * 16, kt * 32, lane);
      hacc[nt] = wmma16(a, b, hacc[nt]);
    }
  }
#pragma unroll
  for (int nt = 0; nt < 8; ++nt) {
    float bias = db1[nt * 16 + ncol];
#pragma unroll
    for (int r = 0; r < 8; ++r) {
      float v = hacc[nt][r] + bias;
      v = v > 0.f ? v : 0.f;
      h2h[(size_t)(m0 + mb + r) * H_ + nt * 16 + ncol] = (_Float16)v;
    }
  }
}

// ---------------------------------------------------------------------------
// Kernel 5: x_hat = h2(2048x128) @ dec_w2(128x20000) + dec_b2
// block tile: 128M x 160N, 256 threads = 8 waves (16 rows each).
// W tile (160x128 f16 = 40KB) staged in LDS; A frags held in registers.
// grid = (2048/128, 20000/160) = (16, 125).
// ---------------------------------------------------------------------------
__global__ __launch_bounds__(256) void gemm_kernel(
    const _Float16* __restrict__ h2h, const _Float16* __restrict__ w2dt,
    const float* __restrict__ db2, float* __restrict__ xhat) {
  __shared__ _Float16 wtile[160 * H_];  // 40 KB, N-major [n][k]
  int lane = threadIdx.x & 31;
  int w    = threadIdx.x >> 5;
  int m0   = blockIdx.x * 128 + w * 16;
  int nb   = blockIdx.y * 160;

  // cooperative, fully-coalesced copy of the contiguous 40KB W slab
  {
    const float4* src = (const float4*)(w2dt + (size_t)nb * H_);
    float4* dst = (float4*)wtile;
#pragma unroll
    for (int i = 0; i < 10; ++i)
      dst[threadIdx.x + i * 256] = src[threadIdx.x + i * 256];
  }
  __syncthreads();

  // A fragments for this wave's 16 rows, K=128 -> 4 frags, kept in VGPRs
  v16h af[4];
#pragma unroll
  for (int kt = 0; kt < 4; ++kt) af[kt] = load_a_frag(h2h, H_, m0, kt * 32, lane);

  int ncol = lane & 15;
  int mb   = (lane >> 4) * 8;

  for (int ntp = 0; ntp < 5; ++ntp) {  // 5 pairs of 16-wide n-tiles = 160
    int nt0 = ntp * 32;
    v8f c0 = zero8(), c1 = zero8();
#pragma unroll
    for (int kt = 0; kt < 4; ++kt) {
      v16h b0 = load_b_frag(wtile, H_, nt0,      kt * 32, lane);
      v16h b1 = load_b_frag(wtile, H_, nt0 + 16, kt * 32, lane);
      c0 = wmma16(af[kt], b0, c0);
      c1 = wmma16(af[kt], b1, c1);
    }
    float bias0 = db2[nb + nt0 + ncol];
    float bias1 = db2[nb + nt0 + 16 + ncol];
#pragma unroll
    for (int r = 0; r < 8; ++r) {
      size_t m = (size_t)(m0 + mb + r);
      xhat[m * G_ + nb + nt0 + ncol]      = c0[r] + bias0;
      xhat[m * G_ + nb + nt0 + 16 + ncol] = c1[r] + bias1;
    }
  }
}

// ---------------------------------------------------------------------------
extern "C" void kernel_launch(void* const* d_in, const int* in_sizes, int n_in,
                              void* d_out, int out_size, void* d_ws,
                              size_t ws_size, hipStream_t stream) {
  const float* x      = (const float*)d_in[0];
  const int*   pidx   = (const int*)d_in[1];
  const float* pw_w   = (const float*)d_in[2];
  const float* pw_b   = (const float*)d_in[3];
  const float* enc_w1 = (const float*)d_in[4];
  const float* enc_b1 = (const float*)d_in[5];
  const float* enc_w2 = (const float*)d_in[6];
  const float* enc_b2 = (const float*)d_in[7];
  const float* dec_w1 = (const float*)d_in[8];
  const float* dec_b1 = (const float*)d_in[9];
  const float* dec_w2 = (const float*)d_in[10];
  const float* dec_b2 = (const float*)d_in[11];

  float* z_out = (float*)d_out;
  float* xhat  = z_out + (size_t)B_ * E_;

  // workspace layout (256B aligned)
  char* ws = (char*)d_ws;
  size_t off = 0;
  auto take = [&](size_t bytes) {
    size_t o = off;
    off = (off + bytes + 255) & ~(size_t)255;
    return o;
  };
  size_t o_w1t  = take((size_t)H_ * P_ * 2);
  size_t o_w2t  = take((size_t)E_ * H_ * 2);
  size_t o_d1t  = take((size_t)H_ * E_ * 2);
  size_t o_w2dt = take((size_t)G_ * H_ * 2);
  size_t o_ah   = take((size_t)B_ * P_ * 2);
  size_t o_h2h  = take((size_t)B_ * H_ * 2);
  size_t o_xt   = take((size_t)G_ * B_ * 4);
  int use_xt = (off <= ws_size) ? 1 : 0;

  _Float16* w1t  = (_Float16*)(ws + o_w1t);
  _Float16* w2t  = (_Float16*)(ws + o_w2t);
  _Float16* d1t  = (_Float16*)(ws + o_d1t);
  _Float16* w2dt = (_Float16*)(ws + o_w2dt);
  _Float16* a_h  = (_Float16*)(ws + o_ah);
  _Float16* h2h  = (_Float16*)(ws + o_h2h);
  float*    x_T  = (float*)(ws + o_xt);

  // 1) weight conversion/transposition (f32 -> f16 N-major)
  {
    int total = H_ * P_ + E_ * H_ + H_ * E_ + G_ * H_;
    int blocks = (total + 255) / 256;
    convert_weights<<<blocks, 256, 0, stream>>>(enc_w1, enc_w2, dec_w1, dec_w2,
                                                w1t, w2t, d1t, w2dt);
  }
  // 2) transpose x for coalesced pathway gather (if workspace permits)
  if (use_xt)
    transpose_x<<<dim3(G_ / 32, B_ / 32), dim3(32, 8), 0, stream>>>(x, x_T);

  // 3) pathway aggregation -> a_h (f16)
  pathway_kernel<<<dim3(P_, B_ / 256), 256, 0, stream>>>(x, x_T, pidx, pw_w,
                                                         pw_b, a_h, use_xt);
  // 4) fused encoder/decoder-head MLP -> z (d_out) and h2 (f16)
  mlp_kernel<<<B_ / 64, 128, 0, stream>>>(a_h, w1t, enc_b1, w2t, enc_b2, d1t,
                                          dec_b1, z_out, h2h);
  // 5) big WMMA GEMM -> x_hat
  gemm_kernel<<<dim3(B_ / 128, G_ / 160), 256, 0, stream>>>(h2h, w2dt, dec_b2,
                                                            xhat);
}
